// LSTMDecoder_31301721653867
// MI455X (gfx1250) — matrix operand order repaired and checked by reading, not compiled, not run on hardware
//
#include <hip/hip_runtime.h>

// ---------------------------------------------------------------------------
// LSTM decoder for MI455X (gfx1250), wave32 + WMMA bf16.
//   B=64, IN=512, H=2048, D=80, T=512, leaky slope 0.04
// Strategy:
//   * one-time: convert weights to bf16 (L2-resident, ~36MB), pad W_ih K to 96
//   * init:  hc0 = tanh(x @ W_h0c0^T + b), x0 = leaky(hc0 @ W_out^T + b_out)
//   * persistent kernel, 128 WGs x 256 thr, 511 steps, 2 grid barriers/step:
//       phase1: gates tile (v_wmma_f32_16x16x32_bf16, K=96 over x + K=2048 over h)
//       phase2: LSTM cell elementwise (c in fp32, h double-buffered bf16)
//       phase3: WGs 0..4 project [h c] @ W_out^T -> x_{t+1} and out[:,t,:]
// ---------------------------------------------------------------------------

typedef __attribute__((ext_vector_type(8)))  float  f32x8;
typedef __attribute__((ext_vector_type(8)))  __bf16 bf16x8;
typedef __attribute__((ext_vector_type(16))) __bf16 bf16x16;

#define NB   64
#define NH   2048
#define NG   8192          // 4*H
#define ND   80
#define NT   512
#define KXP  96            // x/W_ih K padded 80 -> 96 (3 k-tiles of 32)
#define NWG  128
#define SLOPE 0.04f

// ---------------- helpers ----------------

__device__ inline __bf16 to_bf16(float f) {
  union { float f; unsigned u; } v; v.f = f;
  unsigned r = v.u + 0x7fffu + ((v.u >> 16) & 1u);   // round-to-nearest-even
  unsigned short s = (unsigned short)(r >> 16);
  return __builtin_bit_cast(__bf16, s);
}

__device__ inline float sigf(float x)  { return 1.0f / (1.0f + __expf(-x)); }
__device__ inline float leakyf(float x){ return x >= 0.0f ? x : SLOPE * x; }

// A-fragment (16x32, row-major source): lane<16 holds K[0..7]+K[16..23] of row
// (lane&15); lane>=16 holds K[8..15]+K[24..31].  p = &A[row*lda + kbase + (lane>>4)*8]
__device__ inline bf16x16 fragA(const __bf16* p) {
  bf16x8 lo = *(const bf16x8*)p;
  bf16x8 hi = *(const bf16x8*)(p + 16);
  return __builtin_shufflevector(lo, hi, 0,1,2,3,4,5,6,7,8,9,10,11,12,13,14,15);
}

// B-fragment (32x16 = W^T tile from row-major W[n][k]): lane<16 holds
// K[0..7]+K[16..23] of column n=(lane&15); lane>=16 holds K[8..15]+K[24..31].
// p = &W[(nbase + (lane&15))*ldb + kbase + (lane>>4)*8]   (16B aligned)
__device__ inline bf16x16 fragB(const __bf16* p) {
  bf16x8 lo = *(const bf16x8*)p;
  bf16x8 hi = *(const bf16x8*)(p + 16);
  return __builtin_shufflevector(lo, hi, 0,1,2,3,4,5,6,7,8,9,10,11,12,13,14,15);
}

__device__ inline f32x8 wmma_bf16(bf16x16 a, bf16x16 b, f32x8 c) {
  return __builtin_amdgcn_wmma_f32_16x16x32_bf16(
      /*neg_a=*/false, a, /*neg_b=*/false, b,
      /*c_mod=*/(short)0, c, /*reuse_a=*/false, /*reuse_b=*/false);
}

__device__ inline void grid_barrier(unsigned* cnt, unsigned target) {
  __syncthreads();
  __threadfence();                                     // release all WG writes
  if (threadIdx.x == 0) {
    __hip_atomic_fetch_add(cnt, 1u, __ATOMIC_ACQ_REL, __HIP_MEMORY_SCOPE_AGENT);
    while (__hip_atomic_load(cnt, __ATOMIC_ACQUIRE, __HIP_MEMORY_SCOPE_AGENT) < target)
      __builtin_amdgcn_s_sleep(1);
  }
  __syncthreads();
  __threadfence();                                     // acquire side
}

// ---------------- prologue kernels ----------------

__global__ void convert_kernel(const float* __restrict__ Wih,
                               const float* __restrict__ Whh,
                               const float* __restrict__ Wout,
                               const float* __restrict__ bih,
                               const float* __restrict__ bhh,
                               __bf16* wih_p, __bf16* whh_b, __bf16* wout_b,
                               float* bias_g, __bf16* xbuf) {
  const int tid = blockIdx.x * blockDim.x + threadIdx.x;
  const int stride = gridDim.x * blockDim.x;
  for (int i = tid; i < NG * NH; i += stride) whh_b[i] = to_bf16(Whh[i]);
  for (int i = tid; i < NG * KXP; i += stride) {
    int r = i / KXP, c = i - r * KXP;
    wih_p[i] = (c < ND) ? to_bf16(Wih[r * ND + c]) : to_bf16(0.0f);
  }
  for (int i = tid; i < ND * 2 * NH; i += stride) wout_b[i] = to_bf16(Wout[i]);
  for (int i = tid; i < NG; i += stride) bias_g[i] = bih[i] + bhh[i];
  for (int i = tid; i < NB * KXP; i += stride) xbuf[i] = to_bf16(0.0f);
}

// hc0 = tanh(x @ W_h0c0^T + b)  ->  hc0_f32, h0(bf16), c(f32), hc(bf16)
__global__ void init_hc_kernel(const float* __restrict__ x,
                               const float* __restrict__ W,
                               const float* __restrict__ b,
                               float* hc0, float* cbuf, __bf16* h0, __bf16* hcb) {
  const int gid = blockIdx.x * 256 + threadIdx.x;    // 64*4096 exactly
  const int m = gid >> 12;
  const int n = gid & 4095;
  const float* xr = x + m * 512;
  const float* wr = W + n * 512;
  float s = b[n];
  for (int k = 0; k < 512; ++k) s += xr[k] * wr[k];
  const float v = tanhf(s);
  hc0[m * (2 * NH) + n] = v;
  hcb[m * (2 * NH) + n] = to_bf16(v);
  if (n < NH) h0[m * NH + n] = to_bf16(v);
  else        cbuf[m * NH + (n - NH)] = v;
}

// x0 = leaky(hc0 @ W_out^T + b_out) -> out[:,0,:] (f32) and xbuf (bf16)
__global__ void init_x_kernel(const float* __restrict__ hc0,
                              const float* __restrict__ Wout,
                              const float* __restrict__ bout,
                              float* out, __bf16* xbuf) {
  const int gid = blockIdx.x * 256 + threadIdx.x;    // 64*80 exactly
  const int m = gid / ND;
  const int n = gid - m * ND;
  const float* a = hc0 + m * (2 * NH);
  const float* w = Wout + n * (2 * NH);
  float s = bout[n];
  for (int k = 0; k < 2 * NH; ++k) s += a[k] * w[k];
  s = leakyf(s);
  out[m * (NT * ND) + n] = s;                        // t = 0
  xbuf[m * KXP + n] = to_bf16(s);
}

// ---------------- persistent recurrent kernel ----------------

__global__ __launch_bounds__(256)
void lstm_persistent(const __bf16* __restrict__ wih,     // [8192][96]  bf16
                     const __bf16* __restrict__ whh,     // [8192][2048] bf16
                     const __bf16* __restrict__ woutb,   // [80][4096]  bf16
                     const float*  __restrict__ bias_g,  // [8192] b_ih+b_hh
                     const float*  __restrict__ bout,    // [80]
                     __bf16* xbuf,                       // [64][96]  bf16
                     __bf16* h0, __bf16* h1,             // [64][2048] x2
                     float*  cbuf,                       // [64][2048] f32
                     __bf16* hcb,                        // [64][4096] bf16
                     float*  out,                        // [64][512][80]
                     unsigned* barrier_cnt) {
  __shared__ float lds_g[4][64][16];                 // gate tiles, 16 KB
  __shared__ float lds_r[8][16][16];                 // proj K-reduction, 8 KB

  const int wg   = blockIdx.x;                       // 0..127
  const int tid  = threadIdx.x;
  const int w    = tid >> 5;                         // wave 0..7
  const int lane = tid & 31;
  const int l15  = lane & 15;
  const int lhi  = lane >> 4;                        // 0/1

  // phase-1 role: wave -> (gate, M-pair)
  const int gate = w >> 1;                           // 0..3 (i,f,g,o)
  const int m0   = (w & 1) * 32;                     // tiles m0, m0+16
  const int hcol = wg * 16;                          // owned hidden cols
  const int nrow = gate * NH + hcol + l15;           // weight row for B frag
  const float bias = bias_g[gate * NH + hcol + l15];

  for (int t = 1; t < NT; ++t) {
    const __bf16* hread  = (t & 1) ? h0 : h1;
    __bf16*       hwrite = (t & 1) ? h1 : h0;

    // ---- phase 1: gate tiles  D = x@Wih^T + h@Whh^T (+bias) ----
    f32x8 acc0 = {}, acc1 = {};
    {
      const __bf16* pA0 = xbuf + (m0 + l15) * KXP + lhi * 8;
      const __bf16* pA1 = pA0 + 16 * KXP;
      const __bf16* pB  = wih + nrow * KXP + lhi * 8;
      #pragma unroll
      for (int kk = 0; kk < KXP / 32; ++kk) {
        bf16x16 bfr = fragB(pB);
        acc0 = wmma_bf16(fragA(pA0), bfr, acc0);
        acc1 = wmma_bf16(fragA(pA1), bfr, acc1);
        pA0 += 32; pA1 += 32; pB += 32;
      }
    }
    {
      const __bf16* pA0 = hread + (m0 + l15) * NH + lhi * 8;
      const __bf16* pA1 = pA0 + 16 * NH;
      const __bf16* pB  = whh + nrow * NH + lhi * 8;
      #pragma unroll 4
      for (int kk = 0; kk < NH / 32; ++kk) {
        __builtin_prefetch(pB + 256, 0, 1);          // global_prefetch_b8
        bf16x16 bfr = fragB(pB);
        acc0 = wmma_bf16(fragA(pA0), bfr, acc0);
        acc1 = wmma_bf16(fragA(pA1), bfr, acc1);
        pA0 += 32; pA1 += 32; pB += 32;
      }
    }
    #pragma unroll
    for (int v = 0; v < 8; ++v) {                    // C layout: row = v + 8*lhi
      const int r = v + lhi * 8;
      lds_g[gate][m0 + r][l15]      = acc0[v] + bias;
      lds_g[gate][m0 + 16 + r][l15] = acc1[v] + bias;
    }
    __syncthreads();

    // ---- phase 2: LSTM cell elementwise for our 16 columns ----
    #pragma unroll
    for (int e = 0; e < 4; ++e) {
      const int idx = tid + 256 * e;                 // 0..1023 = 64 rows x 16 cols
      const int row = idx >> 4;
      const int col = idx & 15;
      const int n   = hcol + col;
      const float iv = sigf (lds_g[0][row][col]);
      const float fv = sigf (lds_g[1][row][col]);
      const float gv = tanhf(lds_g[2][row][col]);
      const float ov = sigf (lds_g[3][row][col]);
      const float co = cbuf[row * NH + n];
      const float cn = fv * co + iv * gv;
      const float hn = ov * tanhf(cn);
      cbuf  [row * NH + n]            = cn;
      hwrite[row * NH + n]            = to_bf16(hn);
      hcb   [row * (2 * NH) + n]      = to_bf16(hn);
      hcb   [row * (2 * NH) + NH + n] = to_bf16(cn);
    }
    grid_barrier(barrier_cnt, (unsigned)NWG * (unsigned)(2 * t - 1));

    // ---- phase 3: x_{t+1} = leaky([h c] @ W_out^T + b_out), WGs 0..4 ----
    if (wg < 5) {
      const int mt = w & 3;                          // M-tile 0..3
      const int kh = w >> 2;                         // K half 0/1
      f32x8 acc = {};
      const __bf16* pA = hcb   + (mt * 16 + l15) * (2 * NH) + kh * NH + lhi * 8;
      const __bf16* pB = woutb + (wg * 16 + l15) * (2 * NH) + kh * NH + lhi * 8;
      #pragma unroll 4
      for (int kk = 0; kk < NH / 32; ++kk) {
        acc = wmma_bf16(fragA(pA), fragB(pB), acc);
        pA += 32; pB += 32;
      }
      #pragma unroll
      for (int v = 0; v < 8; ++v) lds_r[w][v + lhi * 8][l15] = acc[v];
      __syncthreads();
      if (kh == 0) {
        const int n = wg * 16 + l15;                 // output col < 80
        const float bo = bout[n];
        #pragma unroll
        for (int v = 0; v < 8; ++v) {
          const int r = v + lhi * 8;
          float s = lds_r[w][r][l15] + lds_r[w + 4][r][l15] + bo;
          s = leakyf(s);
          const int m = mt * 16 + r;
          out [m * (NT * ND) + t * ND + n] = s;
          xbuf[m * KXP + n] = to_bf16(s);
        }
      }
    }
    grid_barrier(barrier_cnt, (unsigned)NWG * (unsigned)(2 * t));
  }
}

// ---------------- launch ----------------

extern "C" void kernel_launch(void* const* d_in, const int* in_sizes, int n_in,
                              void* d_out, int out_size, void* d_ws, size_t ws_size,
                              hipStream_t stream) {
  (void)in_sizes; (void)n_in; (void)out_size; (void)ws_size;
  const float* x     = (const float*)d_in[0];
  const float* Wh0c0 = (const float*)d_in[1];
  const float* bh0c0 = (const float*)d_in[2];
  const float* Wih   = (const float*)d_in[3];
  const float* Whh   = (const float*)d_in[4];
  const float* bih   = (const float*)d_in[5];
  const float* bhh   = (const float*)d_in[6];
  const float* Wout  = (const float*)d_in[7];
  const float* bout  = (const float*)d_in[8];
  float* out = (float*)d_out;

  // workspace layout (all offsets 256B aligned), ~36.7 MB total
  char* ws = (char*)d_ws;
  constexpr size_t OFF_CNT  = 0;
  constexpr size_t OFF_XBUF = 256;
  constexpr size_t OFF_BIAS = OFF_XBUF + (size_t)NB * KXP * 2;
  constexpr size_t OFF_C    = OFF_BIAS + (size_t)NG * 4;
  constexpr size_t OFF_H0   = OFF_C    + (size_t)NB * NH * 4;
  constexpr size_t OFF_H1   = OFF_H0   + (size_t)NB * NH * 2;
  constexpr size_t OFF_HC   = OFF_H1   + (size_t)NB * NH * 2;
  constexpr size_t OFF_HC0  = OFF_HC   + (size_t)NB * 2 * NH * 2;
  constexpr size_t OFF_WIH  = OFF_HC0  + (size_t)NB * 2 * NH * 4;
  constexpr size_t OFF_WOUT = OFF_WIH  + (size_t)NG * KXP * 2;
  constexpr size_t OFF_WHH  = OFF_WOUT + (size_t)ND * 2 * NH * 2;

  unsigned* cnt    = (unsigned*)(ws + OFF_CNT);
  __bf16*   xbuf   = (__bf16*)(ws + OFF_XBUF);
  float*    bias_g = (float*) (ws + OFF_BIAS);
  float*    cbuf   = (float*) (ws + OFF_C);
  __bf16*   h0     = (__bf16*)(ws + OFF_H0);
  __bf16*   h1     = (__bf16*)(ws + OFF_H1);
  __bf16*   hcb    = (__bf16*)(ws + OFF_HC);
  float*    hc0    = (float*) (ws + OFF_HC0);
  __bf16*   wih_p  = (__bf16*)(ws + OFF_WIH);
  __bf16*   wout_b = (__bf16*)(ws + OFF_WOUT);
  __bf16*   whh_b  = (__bf16*)(ws + OFF_WHH);

  hipMemsetAsync(ws + OFF_CNT, 0, 256, stream);      // reset grid barrier
  convert_kernel<<<1024, 256, 0, stream>>>(Wih, Whh, Wout, bih, bhh,
                                           wih_p, whh_b, wout_b, bias_g, xbuf);
  init_hc_kernel<<<(NB * 2 * NH) / 256, 256, 0, stream>>>(x, Wh0c0, bh0c0,
                                                          hc0, cbuf, h0, hcb);
  init_x_kernel<<<(NB * ND) / 256, 256, 0, stream>>>(hc0, Wout, bout, out, xbuf);
  lstm_persistent<<<NWG, 256, 0, stream>>>(wih_p, whh_b, wout_b, bias_g, bout,
                                           xbuf, h0, h1, cbuf, hcb, out, cnt);
}
